// PiNet2P5Dot_82102594830599
// MI455X (gfx1250) — matrix operand order repaired and checked by reading, not compiled, not run on hardware
//
#include <hip/hip_runtime.h>
#include <hip/hip_bf16.h>

// ---------------------------------------------------------------------------
// PiNet2P5Dot forward for MI455X (gfx1250, wave32, WMMA).
// Pair-message GEMMs (C=16) mapped onto v_wmma_f32_16x16x32_f16 (f16 A/B,
// f32 accumulate). Atom state lives in f32 in workspace and is L2-resident
// (192MB L2 >> 17MB state). Segment sums via global_atomic_add_f32.
// tanh uses the CDNA5 native V_TANH_F32 transcendental (or a branch-free
// v_exp_f32 fallback) so activation work co-executes with the XDL WMMA pipe
// instead of serializing on exec-mask branches.
// ---------------------------------------------------------------------------

typedef __attribute__((ext_vector_type(16))) _Float16 v16h;
typedef __attribute__((ext_vector_type(8)))  _Float16 v8h;
typedef __attribute__((ext_vector_type(8)))  float    v8f;

#define PI_F 3.14159265358979323846f
#define RC_F 4.0f

// Branch-free tanh: native V_TANH_F32 if the builtin exists, else
// tanh(x) = 1 - 2/(exp(2x)+1) on the hardware exp unit (no EXEC divergence;
// saturates correctly at +/-1 for large |x|).
__device__ __forceinline__ float fast_tanh(float x) {
#if __has_builtin(__builtin_amdgcn_tanhf)
  return __builtin_amdgcn_tanhf(x);
#else
  float e = __expf(2.f * x);
  return 1.f - 2.f / (e + 1.f);
#endif
}

__device__ __forceinline__ v16h cat16(v8h lo, v8h hi) {
  return __builtin_shufflevector(lo, hi, 0,1,2,3,4,5,6,7,8,9,10,11,12,13,14,15);
}

// Convert 8 consecutive f32 (16B-aligned) to 8 f16.
__device__ __forceinline__ v8h cvt8(const float* p) {
  float4 a = ((const float4*)p)[0];
  float4 b = ((const float4*)p)[1];
  v8h r;
  r[0] = (_Float16)a.x; r[1] = (_Float16)a.y; r[2] = (_Float16)a.z; r[3] = (_Float16)a.w;
  r[4] = (_Float16)b.x; r[5] = (_Float16)b.y; r[6] = (_Float16)b.z; r[7] = (_Float16)b.w;
  return r;
}

// A-tile (16x32 f16, row-major in LDS, row stride 32 halves).
// Lane (m = lane&15, h = lane>>4) holds K = [8h,8h+8) then [16+8h,16+8h+8).
__device__ __forceinline__ v16h load_at(const _Float16* Ab, int lane) {
  int m = lane & 15, h = lane >> 4;
  const v8h* p0 = (const v8h*)(Ab + m*32 + 8*h);
  const v8h* p1 = (const v8h*)(Ab + m*32 + 16 + 8*h);
  return cat16(*p0, *p1);
}

// B-tile (K=32 x N=16), staged transposed: Bt[n*32 + k].
// Lane (n = lane&15, h) holds K = [8h,8h+8) then [16+8h,16+8h+8) of column n.
__device__ __forceinline__ v16h load_bt(const _Float16* Bt, int lane) {
  int n = lane & 15, h = lane >> 4;
  const v8h* p0 = (const v8h*)(Bt + n*32 + 8*h);
  const v8h* p1 = (const v8h*)(Bt + n*32 + 16 + 8*h);
  return cat16(*p0, *p1);
}

__device__ __forceinline__ v8f wmma_f16(v16h a, v16h b, v8f c) {
  // D = A(16x32) x B(32x16) + C, f32 accumulate (probe-confirmed builtin).
  return __builtin_amdgcn_wmma_f32_16x16x32_f16(false, a, false, b, (short)0, c, false, false);
}

// ---------------------------------------------------------------------------
// Small utility kernels
// ---------------------------------------------------------------------------
__global__ void k_zero(float* p, size_t n) {
  size_t i = (size_t)blockIdx.x * 256 + threadIdx.x;
  if (i < n) p[i] = 0.f;
}

__global__ void k_reduce_ss(const float* __restrict__ d, int n, float* ssum) {
  __shared__ float red[256];
  int i = blockIdx.x * 256 + threadIdx.x;
  float v = (i < n) ? d[i] * d[i] : 0.f;
  red[threadIdx.x] = v;
  __syncthreads();
  for (int s = 128; s > 0; s >>= 1) {
    if (threadIdx.x < s) red[threadIdx.x] += red[threadIdx.x + s];
    __syncthreads();
  }
  if (threadIdx.x == 0) atomicAdd(ssum, red[0]);
}

// ndiff = diff / ||diff||_F (global scalar norm, matching jnp.linalg.norm(diff)),
// diff_p5 quadratic features, cosine cutoff basis powers.
__global__ void k_pair_pre(const float* __restrict__ diff, const float* __restrict__ dist,
                           const float* __restrict__ ssum,
                           float* ndw, float* dpw, float* basw, int P) {
  int p = blockIdx.x * 256 + threadIdx.x;
  if (p >= P) return;
  float inv = rsqrtf(*ssum);
  float x = diff[p*3+0] * inv, y = diff[p*3+1] * inv, z = diff[p*3+2] * inv;
  ndw[p*3+0] = x; ndw[p*3+1] = y; ndw[p*3+2] = z;
  float x2 = x*x, y2 = y*y, z2 = z*z;
  const float th = 1.f/3.f;
  dpw[p*5+0] = 2.f*th*x2 - th*y2 - th*z2;
  dpw[p*5+1] = 2.f*th*y2 - th*x2 - th*z2;
  dpw[p*5+2] = x*y;
  dpw[p*5+3] = x*z;
  dpw[p*5+4] = y*z;
  float fc = 0.5f * (cosf(PI_F * dist[p] / RC_F) + 1.f);
  float f = fc;
  basw[p*4+0] = f; f *= fc;
  basw[p*4+1] = f; f *= fc;
  basw[p*4+2] = f; f *= fc;
  basw[p*4+3] = f;
}

// p1 state = padded one-hot over {1,6,7,8}; p3/p5 = 0; record type index.
__global__ void k_atom_init(const int* __restrict__ elems, float* p1s, float* p3s,
                            float* p5s, int* tidx, int N) {
  int a = blockIdx.x * 256 + threadIdx.x;
  if (a >= N) return;
  int e = elems[a];
  int t = (e == 1) ? 0 : (e == 6) ? 1 : (e == 7) ? 2 : 3;
  tidx[a] = t;
  for (int c = 0; c < 16; ++c) p1s[(size_t)a*16 + c] = (c == t) ? 1.f : 0.f;
  for (int k = 0; k < 48; ++k) p3s[(size_t)a*48 + k] = 0.f;
  for (int k = 0; k < 80; ++k) p5s[(size_t)a*80 + k] = 0.f;
}

// Pad depth-0 pi1[0].w (8x16) into 32x16: rows 0..3 <- W[0:4], rows 16..19 <- W[4:8].
__global__ void k_pad_w0(const float* __restrict__ w8, float* wpad) {
  int t = blockIdx.x * 256 + threadIdx.x;
  if (t >= 512) return;
  int k = t >> 4, n = t & 15;
  float v = 0.f;
  if (k < 4) v = w8[k*16 + n];
  else if (k >= 16 && k < 20) v = w8[(k-12)*16 + n];
  wpad[k*16 + n] = v;
}

// ---------------------------------------------------------------------------
// Pair kernel: 8 waves x 16 pairs per 256-thread block. 19 WMMAs per tile.
// ---------------------------------------------------------------------------
template<bool FIRST>
__global__ __launch_bounds__(256)
void k_pairs(const int* __restrict__ ind2,
             const float* __restrict__ p1g, const float* __restrict__ p3g,
             const float* __restrict__ p5g,
             const float* __restrict__ ndw, const float* __restrict__ dpw,
             const float* __restrict__ basw,
             float* p1n, float* p3n, float* p5n,
             const float* __restrict__ Wpi10, const float* __restrict__ bpi10,
             const float* __restrict__ Wpi11, const float* __restrict__ bpi11,
             const float* __restrict__ Wii0,  const float* __restrict__ Wii1,
             const float* __restrict__ W3i,   const float* __restrict__ W3j,
             const float* __restrict__ W5i,   const float* __restrict__ W5j,
             int P) {
  // Weight tiles in LDS, transposed (Bt[n*32+k]) and zero-padded to K=32.
  __shared__ __align__(16) _Float16 sB1t[512];      // [p1i|p1j](32) -> 16
  __shared__ __align__(16) _Float16 sB2t[4*512];    // 16 -> 64 (4 N-tiles)
  __shared__ __align__(16) _Float16 sBii0[512];     // 16 -> 16
  __shared__ __align__(16) _Float16 sBii1[5*512];   // 16 -> 80 (5 N-tiles)
  __shared__ __align__(16) _Float16 sB3t[512];      // [p3i|p3j](32) -> 16
  __shared__ __align__(16) _Float16 sB5t[512];      // [p5i|p5j](32) -> 16
  __shared__ __align__(16) _Float16 sAb[8][512];    // per-wave A staging 16x32
  __shared__ __align__(16) float    sX[8][1024];    // per-wave X (16x64 f32)

  const int tid = threadIdx.x;
  for (int t = tid; t < 512; t += 256) {
    int k = t >> 4, n = t & 15;
    sB1t[n*32 + k] = (_Float16)Wpi10[k*16 + n];
  }
  for (int t = tid; t < 2048; t += 256) {
    int g = t >> 9, r = t & 511, k = r >> 4, n = r & 15;
    sB2t[g*512 + n*32 + k] = (k < 16) ? (_Float16)Wpi11[k*64 + 16*g + n] : (_Float16)0.f;
  }
  for (int t = tid; t < 512; t += 256) {
    int k = t >> 4, n = t & 15;
    sBii0[n*32 + k] = (k < 16) ? (_Float16)Wii0[k*16 + n] : (_Float16)0.f;
  }
  for (int t = tid; t < 2560; t += 256) {
    int g = t / 512, r = t % 512, k = r >> 4, n = r & 15;
    sBii1[g*512 + n*32 + k] = (k < 16) ? (_Float16)Wii1[k*80 + 16*g + n] : (_Float16)0.f;
  }
  for (int t = tid; t < 512; t += 256) {
    int k = t >> 4, n = t & 15;
    _Float16 v3 = (_Float16)0.f, v5 = (_Float16)0.f;
    if (!FIRST) {
      v3 = (k < 16) ? (_Float16)W3i[k*16 + n] : (_Float16)W3j[(k-16)*16 + n];
      v5 = (k < 16) ? (_Float16)W5i[k*16 + n] : (_Float16)W5j[(k-16)*16 + n];
    }
    sB3t[n*32 + k] = v3;
    sB5t[n*32 + k] = v5;
  }
  __syncthreads();

  const int lane = threadIdx.x & 31;
  const int wave = threadIdx.x >> 5;
  const int ln = lane & 15;     // A row / B,D column
  const int hf = lane >> 4;
  long tile = (long)blockIdx.x * 8 + wave;
  long base = tile * 16;
  if (base >= P) return;        // wave-uniform; active waves keep EXEC all-ones

  _Float16* Ab = sAb[wave];
  float*    Xb = sX[wave];
  // Zero the K=16..31 pad of the A buffer once (only cols 0..15 written later).
  #pragma unroll
  for (int t = 0; t < 8; ++t) Ab[ln*32 + 16 + 8*hf + t] = (_Float16)0.f;

  const int2 ind = ((const int2*)ind2)[base + ln];
  v8f zacc = {};

  // ---- pi1 layer 0: H = tanh([p1[i]|p1[j]] @ Wstack + b0) ----
  {
    const float* r1i = p1g + (size_t)ind.x * 16 + 8*hf;
    const float* r1j = p1g + (size_t)ind.y * 16 + 8*hf;
    v8f acc = wmma_f16(cat16(cvt8(r1i), cvt8(r1j)), load_bt(sB1t, lane), zacc);
    float bb = bpi10[ln];
    #pragma unroll
    for (int v = 0; v < 8; ++v) Ab[(v + 8*hf)*32 + ln] = (_Float16)fast_tanh(acc[v] + bb);
  }

  // ---- pi1 layer 1: X = tanh(H @ W2 + b1), 16x64 ----
  {
    v16h aH = load_at(Ab, lane);
    #pragma unroll
    for (int g = 0; g < 4; ++g) {
      v8f xg = wmma_f16(aH, load_bt(sB2t + g*512, lane), zacc);
      float bg = bpi11[16*g + ln];
      #pragma unroll
      for (int v = 0; v < 8; ++v) Xb[(v + 8*hf)*64 + 16*g + ln] = fast_tanh(xg[v] + bg);
    }
  }

  // ---- basis contraction: i1[m][c] = sum_b X[m][4c+b]*basis[m][b] ----
  {
    const float* bp = basw + (size_t)(base + ln) * 4;
    float b0 = bp[0], b1v = bp[1], b2 = bp[2], b3 = bp[3];
    #pragma unroll
    for (int cc = 0; cc < 8; ++cc) {
      int c = 8*hf + cc;
      const float* xp = Xb + ln*64 + 4*c;
      Ab[ln*32 + c] = (_Float16)(xp[0]*b0 + xp[1]*b1v + xp[2]*b2 + xp[3]*b3);
    }
  }

  // ---- ii1 layer 0: J = tanh(i1 @ Wii0) ----
  {
    v8f j0 = wmma_f16(load_at(Ab, lane), load_bt(sBii0, lane), zacc);
    #pragma unroll
    for (int v = 0; v < 8; ++v) Ab[(v + 8*hf)*32 + ln] = (_Float16)fast_tanh(j0[v]);
  }

  // ---- ii1 layer 1: 5 tiles (i1_1 .. i1_5), tanh ----
  v16h aJ = load_at(Ab, lane);
  v8f T0 = wmma_f16(aJ, load_bt(sBii1 + 0*512, lane), zacc);
  v8f T1 = wmma_f16(aJ, load_bt(sBii1 + 1*512, lane), zacc);
  v8f T2 = wmma_f16(aJ, load_bt(sBii1 + 2*512, lane), zacc);
  v8f T3 = wmma_f16(aJ, load_bt(sBii1 + 3*512, lane), zacc);
  v8f T4 = wmma_f16(aJ, load_bt(sBii1 + 4*512, lane), zacc);
  #pragma unroll
  for (int v = 0; v < 8; ++v) {
    T0[v] = fast_tanh(T0[v]); T1[v] = fast_tanh(T1[v]); T2[v] = fast_tanh(T2[v]);
    T3[v] = fast_tanh(T3[v]); T4[v] = fast_tanh(T4[v]);
  }

  // ---- p1_new scatter (segment sum over i) ----
  #pragma unroll
  for (int v = 0; v < 8; ++v) {
    int M = v + 8*hf;
    int ai = __shfl(ind.x, M);
    atomicAdd(p1n + (size_t)ai*16 + ln, T0[v]);
  }

  // geometry rows for this pair (row ln); broadcast across lanes via shuffles
  float nd[3], dp[5];
  {
    const float* q = ndw + (size_t)(base + ln) * 3;
    nd[0] = q[0]; nd[1] = q[1]; nd[2] = q[2];
    const float* r = dpw + (size_t)(base + ln) * 5;
    dp[0] = r[0]; dp[1] = r[1]; dp[2] = r[2]; dp[3] = r[3]; dp[4] = r[4];
  }

  // ---- i3 = (Wi p3[i] + Wj p3[j]) * i1_4 + ndiff x i1_2, scatter ----
  #pragma unroll
  for (int x = 0; x < 3; ++x) {
    v8f d3 = zacc;
    if (!FIRST) {
      const float* r3i = p3g + ((size_t)ind.x*3 + x)*16 + 8*hf;
      const float* r3j = p3g + ((size_t)ind.y*3 + x)*16 + 8*hf;
      d3 = wmma_f16(cat16(cvt8(r3i), cvt8(r3j)), load_bt(sB3t, lane), zacc);
    }
    #pragma unroll
    for (int v = 0; v < 8; ++v) {
      int M = v + 8*hf;
      float ndx = __shfl(nd[x], M);
      int ai = __shfl(ind.x, M);
      float val = ndx * T1[v];
      if (!FIRST) val += d3[v] * T3[v];
      atomicAdd(p3n + ((size_t)ai*3 + x)*16 + ln, val);
    }
  }

  // ---- i5 = (Wi p5[i] + Wj p5[j]) * i1_5 + diff_p5 x i1_3, scatter ----
  #pragma unroll
  for (int x = 0; x < 5; ++x) {
    v8f d5 = zacc;
    if (!FIRST) {
      const float* r5i = p5g + ((size_t)ind.x*5 + x)*16 + 8*hf;
      const float* r5j = p5g + ((size_t)ind.y*5 + x)*16 + 8*hf;
      d5 = wmma_f16(cat16(cvt8(r5i), cvt8(r5j)), load_bt(sB5t, lane), zacc);
    }
    #pragma unroll
    for (int v = 0; v < 8; ++v) {
      int M = v + 8*hf;
      float dpx = __shfl(dp[x], M);
      int ai = __shfl(ind.x, M);
      float val = dpx * T2[v];
      if (!FIRST) val += d5[v] * T4[v];
      atomicAdd(p5n + ((size_t)ai*5 + x)*16 + ln, val);
    }
  }
}

// ---------------------------------------------------------------------------
// Per-atom head: dots, pp1 MLP, pp3/pp5, output head, residual state update.
// 30k atoms -> ~2 GFLOP total; one thread per atom is plenty.
// ---------------------------------------------------------------------------
struct AtomW {
  const float *d1i, *d1j, *d2i, *d2j;
  const float *pp10b, *pp10w, *pp11b, *pp11w, *pp3w, *pp5w;
  const float *f0b, *f0w, *f1b, *f1w, *outw;
};

template<bool FIRST>
__global__ void k_atoms(int N,
                        const float* __restrict__ p1n, const float* __restrict__ p3n,
                        const float* __restrict__ p5n,
                        float* p1s, float* p3s, float* p5s,
                        const int* __restrict__ tidx, const float* __restrict__ res1w,
                        AtomW w, float* outp) {
  int a = blockIdx.x * blockDim.x + threadIdx.x;
  if (a >= N) return;

  float x1[16];
  for (int c = 0; c < 16; ++c) x1[c] = p1n[(size_t)a*16 + c];

  float dot5[16];
  for (int r = 0; r < 16; ++r) dot5[r] = 0.f;
  for (int x = 0; x < 5; ++x) {
    float row[16];
    for (int c = 0; c < 16; ++c) row[c] = p5n[((size_t)a*5 + x)*16 + c];
    for (int r = 0; r < 16; ++r) {
      float u = 0.f, vv = 0.f;
      for (int c = 0; c < 16; ++c) { u += row[c]*w.d1i[c*16+r]; vv += row[c]*w.d1j[c*16+r]; }
      dot5[r] += u * vv;
    }
  }
  float dot3[16];
  for (int r = 0; r < 16; ++r) dot3[r] = 0.f;
  for (int x = 0; x < 3; ++x) {
    float row[16];
    for (int c = 0; c < 16; ++c) row[c] = p3n[((size_t)a*3 + x)*16 + c];
    for (int r = 0; r < 16; ++r) {
      float u = 0.f, vv = 0.f;
      for (int c = 0; c < 16; ++c) { u += row[c]*w.d2i[c*16+r]; vv += row[c]*w.d2j[c*16+r]; }
      dot3[r] += u * vv;
    }
  }

  float t1[16];
  for (int r = 0; r < 16; ++r) {
    float s = w.pp10b[r];
    for (int k = 0; k < 16; ++k) s += dot5[k] * w.pp10w[k*16 + r];
    for (int k = 0; k < 16; ++k) s += dot3[k] * w.pp10w[(16+k)*16 + r];
    for (int k = 0; k < 16; ++k) s += x1[k]   * w.pp10w[(32+k)*16 + r];
    t1[r] = fast_tanh(s);
  }
  float t2[48];
  for (int r = 0; r < 48; ++r) {
    float s = w.pp11b[r];
    for (int k = 0; k < 16; ++k) s += t1[k] * w.pp11w[k*48 + r];
    t2[r] = fast_tanh(s);
  }

  for (int x = 0; x < 3; ++x) {
    float q[16];
    for (int c = 0; c < 16; ++c) q[c] = p3n[((size_t)a*3 + x)*16 + c] * t2[16 + c];
    for (int r = 0; r < 16; ++r) {
      float s = 0.f;
      for (int c = 0; c < 16; ++c) s += q[c] * w.pp3w[c*16 + r];
      size_t off = ((size_t)a*3 + x)*16 + r;
      p3s[off] = (FIRST ? 0.f : p3s[off]) + s;
    }
  }
  for (int x = 0; x < 5; ++x) {
    float q[16];
    for (int c = 0; c < 16; ++c) q[c] = p5n[((size_t)a*5 + x)*16 + c] * t2[32 + c];
    for (int r = 0; r < 16; ++r) {
      float s = 0.f;
      for (int c = 0; c < 16; ++c) s += q[c] * w.pp5w[c*16 + r];
      size_t off = ((size_t)a*5 + x)*16 + r;
      p5s[off] = (FIRST ? 0.f : p5s[off]) + s;
    }
  }
  {
    int t = FIRST ? tidx[a] : 0;
    for (int r = 0; r < 16; ++r) {
      size_t off = (size_t)a*16 + r;
      float basev = FIRST ? res1w[t*16 + r] : p1s[off];
      p1s[off] = basev + t2[r];   // p1t1 = t2[0:16]
    }
  }
  float h1[16];
  for (int r = 0; r < 16; ++r) {
    float s = w.f0b[r];
    for (int k = 0; k < 16; ++k) s += t2[k] * w.f0w[k*16 + r];
    h1[r] = fast_tanh(s);
  }
  float o = 0.f;
  for (int r = 0; r < 16; ++r) {
    float s = w.f1b[r];
    for (int k = 0; k < 16; ++k) s += h1[k] * w.f1w[k*16 + r];
    o += fast_tanh(s) * w.outw[r];
  }
  outp[a] = (FIRST ? 0.f : outp[a]) + o;
}

// ---------------------------------------------------------------------------
// Launch. Param leaves follow JAX pytree flatten order:
//   d_in[0..4] = ind_1, ind_2, elems, diff, dist
//   per block d (base = 5 + 25*d): dot1_wi, dot1_wj, dot2_wi, dot2_wj,
//     ii1[0].w, ii1[1].w, out_ff[0].b/w, out_ff[1].b/w, out_w.w,
//     pi1[0].b/w, pi1[1].b/w, pi3_wi, pi3_wj, pi5_wi, pi5_wj,
//     pp1[0].b/w, pp1[1].b/w, pp3.w, pp5.w
//   d_in[105..107] = res1.w, res3.w, res5.w
// ---------------------------------------------------------------------------
extern "C" void kernel_launch(void* const* d_in, const int* in_sizes, int n_in,
                              void* d_out, int out_size, void* d_ws, size_t ws_size,
                              hipStream_t stream) {
  const int*   ind2  = (const int*)d_in[1];
  const int*   elems = (const int*)d_in[2];
  const float* diff  = (const float*)d_in[3];
  const float* dist  = (const float*)d_in[4];
  const int P = in_sizes[1] / 2;
  const int N = in_sizes[2];
  (void)n_in; (void)ws_size; (void)out_size;

  auto Wp = [&](int i) { return (const float*)d_in[i]; };

  float* ws = (float*)d_ws;
  size_t o = 0;
  float* ndw  = ws + o; o += (size_t)3 * P;
  float* dpw  = ws + o; o += (size_t)5 * P;
  float* basw = ws + o; o += (size_t)4 * P;
  float* p1s  = ws + o; o += (size_t)16 * N;
  float* p3s  = ws + o; o += (size_t)48 * N;
  float* p5s  = ws + o; o += (size_t)80 * N;
  float* p1n  = ws + o; o += (size_t)16 * N;
  float* p3n  = ws + o; o += (size_t)48 * N;
  float* p5n  = ws + o; o += (size_t)80 * N;
  int*   tidx = (int*)(ws + o); o += (size_t)N;
  float* wpad = ws + o; o += 512;
  float* ssum = ws + o; o += 1;
  float* outp = (float*)d_out;

  k_zero<<<1, 256, 0, stream>>>(ssum, 1);
  k_reduce_ss<<<(3*P + 255) / 256, 256, 0, stream>>>(diff, 3*P, ssum);
  k_pair_pre<<<(P + 255) / 256, 256, 0, stream>>>(diff, dist, ssum, ndw, dpw, basw, P);
  k_atom_init<<<(N + 255) / 256, 256, 0, stream>>>(elems, p1s, p3s, p5s, tidx, N);
  k_pad_w0<<<2, 256, 0, stream>>>(Wp(5 + 12), wpad);   // block0 pi1[0].w (8x16)

  const int tiles = (P + 15) / 16;
  const int pairBlocks = (tiles + 7) / 8;
  const size_t accElems = (size_t)144 * N;   // p1n|p3n|p5n contiguous
  const int atomBlocks = (N + 255) / 256;

  for (int d = 0; d < 4; ++d) {
    const int b = 5 + 25 * d;
    k_zero<<<(int)((accElems + 255) / 256), 256, 0, stream>>>(p1n, accElems);

    const float* Wpi10 = (d == 0) ? wpad : Wp(b + 12);
    if (d == 0) {
      k_pairs<true><<<pairBlocks, 256, 0, stream>>>(
          ind2, p1s, p3s, p5s, ndw, dpw, basw, p1n, p3n, p5n,
          Wpi10, Wp(b+11), Wp(b+14), Wp(b+13), Wp(b+4), Wp(b+5),
          Wp(b+15), Wp(b+16), Wp(b+17), Wp(b+18), P);
    } else {
      k_pairs<false><<<pairBlocks, 256, 0, stream>>>(
          ind2, p1s, p3s, p5s, ndw, dpw, basw, p1n, p3n, p5n,
          Wpi10, Wp(b+11), Wp(b+14), Wp(b+13), Wp(b+4), Wp(b+5),
          Wp(b+15), Wp(b+16), Wp(b+17), Wp(b+18), P);
    }

    AtomW aw;
    aw.d1i = Wp(b+0);  aw.d1j = Wp(b+1);  aw.d2i = Wp(b+2);  aw.d2j = Wp(b+3);
    aw.pp10b = Wp(b+19); aw.pp10w = Wp(b+20); aw.pp11b = Wp(b+21); aw.pp11w = Wp(b+22);
    aw.pp3w = Wp(b+23); aw.pp5w = Wp(b+24);
    aw.f0b = Wp(b+6); aw.f0w = Wp(b+7); aw.f1b = Wp(b+8); aw.f1w = Wp(b+9);
    aw.outw = Wp(b+10);

    if (d == 0) {
      k_atoms<true><<<atomBlocks, 256, 0, stream>>>(
          N, p1n, p3n, p5n, p1s, p3s, p5s, tidx, Wp(105), aw, outp);
    } else {
      k_atoms<false><<<atomBlocks, 256, 0, stream>>>(
          N, p1n, p3n, p5n, p1s, p3s, p5s, tidx, Wp(105), aw, outp);
    }
  }
}
